// CodeBook_37349035606535
// MI455X (gfx1250) — matrix operand order repaired and checked by reading, not compiled, not run on hardware
//
#include <hip/hip_runtime.h>

typedef __attribute__((ext_vector_type(16))) __bf16 v16bf;
typedef __attribute__((ext_vector_type(8)))  __bf16 bf16x8;
typedef __attribute__((ext_vector_type(8)))  float  v8f;

namespace {
constexpr int Bb = 4, Dd = 256, Nn = 4096, Kk = 8192;
constexpr int NT = 64;        // n-rows per block (4 subtiles x 16)
constexpr int KC = 64;        // k chunk per LDS stage (4 k-tiles of 16)
constexpr int STRIDE = 264;   // padded bf16 row stride (33*8 -> bank spread)
constexpr int NCH = Kk / KC;  // 128 chunks
}

// Raw 32-bit LDS byte offset of a shared-memory pointer (for async-to-LDS VDST).
template <typename T>
__device__ __forceinline__ unsigned lds_off_of(T* p) {
  return (unsigned)(unsigned long long)(__attribute__((address_space(3))) T*)p;
}

// CDNA5 async global->LDS copy, 16 bytes per lane, tracked by ASYNCcnt.
__device__ __forceinline__ void async_copy_b128(unsigned lds_dst, const void* gsrc) {
  asm volatile("global_load_async_to_lds_b128 %0, %1, off"
               :: "v"(lds_dst), "v"(gsrc) : "memory");
}

// ---------------------------------------------------------------------------
// Kernel 1: e (f32) -> bf16 workspace + e2[k] = sum_d e[k,d]^2
// ---------------------------------------------------------------------------
__global__ __launch_bounds__(256) void vq_prep_kernel(
    const float* __restrict__ e, __bf16* __restrict__ ebf,
    float* __restrict__ e2) {
  __shared__ float sred[256];
  const int k = blockIdx.x;
  const int tid = threadIdx.x;
  float v = e[(size_t)k * Dd + tid];
  ebf[(size_t)k * Dd + tid] = (__bf16)v;
  sred[tid] = v * v;
  __syncthreads();
  for (int s = 128; s > 0; s >>= 1) {
    if (tid < s) sred[tid] += sred[tid + s];
    __syncthreads();
  }
  if (tid == 0) e2[k] = sred[0];
}

// ---------------------------------------------------------------------------
// Kernel 2: fused GEMM + argmin.  score[n,k] = e2[k] - 2 * <x_n, e_k>
// Double-buffered async-to-LDS chunk streaming; deep-unrolled WMMA body.
// ---------------------------------------------------------------------------
__global__ __launch_bounds__(256)
__attribute__((amdgpu_waves_per_eu(2)))
void vq_argmin_kernel(
    const float* __restrict__ x, const __bf16* __restrict__ ebf,
    const float* __restrict__ e2, int* __restrict__ codes) {
  __shared__ __align__(16) __bf16 tile[2][NT * STRIDE];  // x-tile, then e ping/pong
  __shared__ float red_val[4][16][2];
  __shared__ int   red_idx[4][16][2];

  const int tid  = threadIdx.x;
  const int wave = tid >> 5;
  const int lane = tid & 31;
  const int col  = lane & 15;   // matrix column within 16x16 tile
  const int h    = lane >> 4;   // lane half
  const int nsub = wave & 3;    // 16-row n-subtile owned by this wave
  const int kh   = wave >> 2;   // half of the k-chunk owned by this wave

  const int b  = blockIdx.x / (Nn / NT);
  const int n0 = (blockIdx.x % (Nn / NT)) * NT;

  // ---- stage x[b, :, n0..n0+63] into LDS buf0 as bf16 rows [i][d] ----
  const float* xb = x + (size_t)b * Dd * Nn;
  #pragma unroll 4
  for (int it = 0; it < 64; ++it) {
    const int d = it * 4 + (tid >> 6);
    const int i = tid & 63;
    tile[0][i * STRIDE + d] = (__bf16)xb[(size_t)d * Nn + n0 + i];
  }
  __syncthreads();

  // ---- A fragments to registers: 16 rows x 256 d, reused for all K ----
  const int m = nsub * 16 + col;
  v16bf afrag[8];
  #pragma unroll
  for (int s = 0; s < 8; ++s) {
    union { v16bf v; bf16x8 p[2]; } u;
    u.p[0] = *(const bf16x8*)&tile[0][m * STRIDE + s * 32 + 8 * h];
    u.p[1] = *(const bf16x8*)&tile[0][m * STRIDE + s * 32 + 16 + 8 * h];
    afrag[s] = u.v;
  }
  __syncthreads();  // x-tile consumed; buffers free for e-chunk ping/pong

  const unsigned lbase[2] = { lds_off_of(&tile[0][0]), lds_off_of(&tile[1][0]) };
  const int cr = tid >> 2;      // this thread's chunk row
  const int cq = tid & 3;       // quarter of the row
  const unsigned ldst_off = (unsigned)((cr * STRIDE + cq * 64) * sizeof(__bf16));

  // kick off chunk 0 into buf0
  {
    const __bf16* src = ebf + (size_t)cr * Dd + cq * 64;
    #pragma unroll
    for (int j = 0; j < 8; ++j)
      async_copy_b128(lbase[0] + ldst_off + j * 16, src + j * 8);
  }

  float bval[8];
  int   bidx[8];
  #pragma unroll
  for (int r = 0; r < 8; ++r) { bval[r] = 3.4e38f; bidx[r] = 0; }

  for (int c = 0; c < NCH; ++c) {
    const int cur = c & 1;
    if (c + 1 < NCH) {
      // overlap: issue next chunk's async loads into the other buffer
      const __bf16* src = ebf + (size_t)((c + 1) * KC + cr) * Dd + cq * 64;
      const unsigned dst = lbase[cur ^ 1] + ldst_off;
      #pragma unroll
      for (int j = 0; j < 8; ++j)
        async_copy_b128(dst + j * 16, src + j * 8);
      // async loads complete in order: <=8 outstanding => chunk c done
      asm volatile("s_wait_asynccnt 0x8" ::: "memory");
    } else {
      asm volatile("s_wait_asynccnt 0x0" ::: "memory");
    }
    __syncthreads();

    const __bf16* buf = &tile[cur][0];
    const int kc  = c * KC;
    const int kt0 = kh * 2;
    const __bf16* bp0 = buf + (kt0 * 16 + col) * STRIDE + 8 * h;
    const __bf16* bp1 = bp0 + 16 * STRIDE;

    // preload BOTH k-tiles' B fragments (32x ds_load_b128), then 16 WMMAs
    v16bf bfr0[8], bfr1[8];
    #pragma unroll
    for (int s = 0; s < 8; ++s) {
      union { v16bf v; bf16x8 p[2]; } u0, u1;
      u0.p[0] = *(const bf16x8*)(bp0 + s * 32);
      u0.p[1] = *(const bf16x8*)(bp0 + s * 32 + 16);
      u1.p[0] = *(const bf16x8*)(bp1 + s * 32);
      u1.p[1] = *(const bf16x8*)(bp1 + s * 32 + 16);
      bfr0[s] = u0.v;
      bfr1[s] = u1.v;
    }
    v8f acc0 = {}, acc1 = {};
    #pragma unroll
    for (int s = 0; s < 8; ++s) {   // two independent chains -> XDL pipelining
      acc0 = __builtin_amdgcn_wmma_f32_16x16x32_bf16(
          false, afrag[s], false, bfr0[s], (short)0, acc0, false, false);
      acc1 = __builtin_amdgcn_wmma_f32_16x16x32_bf16(
          false, afrag[s], false, bfr1[s], (short)0, acc1, false, false);
    }

    const int   ki0 = kc + kt0 * 16 + col;
    const float e20 = e2[ki0];
    const float e21 = e2[ki0 + 16];
    #pragma unroll
    for (int r = 0; r < 8; ++r) {
      const float s0 = fmaf(-2.0f, acc0[r], e20);
      if (s0 < bval[r]) { bval[r] = s0; bidx[r] = ki0; }
      const float s1 = fmaf(-2.0f, acc1[r], e21);
      if (s1 < bval[r]) { bval[r] = s1; bidx[r] = ki0 + 16; }
    }
    __syncthreads();  // chunk consumed before its buffer is overwritten
  }

  // ---- argmin across the 16 columns (stays inside each 16-lane group) ----
  #pragma unroll
  for (int mask = 1; mask < 16; mask <<= 1) {
    #pragma unroll
    for (int r = 0; r < 8; ++r) {
      const float ov = __shfl_xor(bval[r], mask, 32);
      const int   oi = __shfl_xor(bidx[r], mask, 32);
      if (ov < bval[r] || (ov == bval[r] && oi < bidx[r])) {
        bval[r] = ov; bidx[r] = oi;
      }
    }
  }
  if (col == 0) {
    #pragma unroll
    for (int r = 0; r < 8; ++r) {
      red_val[nsub][h * 8 + r][kh] = bval[r];
      red_idx[nsub][h * 8 + r][kh] = bidx[r];
    }
  }
  __syncthreads();

  if (tid < 64) {
    const int ns = tid >> 4, row = tid & 15;
    const float v0 = red_val[ns][row][0], v1 = red_val[ns][row][1];
    const int   i0 = red_idx[ns][row][0], i1 = red_idx[ns][row][1];
    const int code = (v1 < v0 || (v1 == v0 && i1 < i0)) ? i1 : i0;
    codes[(size_t)b * Nn + n0 + ns * 16 + row] = code;
  }
}

// ---------------------------------------------------------------------------
// Kernel 3: decode  out[b,d,n] = e[codes[b,n], d]  (coalesced via LDS)
// ---------------------------------------------------------------------------
__global__ __launch_bounds__(256) void vq_decode_kernel(
    const float* __restrict__ e, const int* __restrict__ codes,
    float* __restrict__ out) {
  constexpr int NTD = 32;
  __shared__ float rows[NTD][Dd + 1];  // stride 257 -> conflict-free transpose
  __shared__ int scodes[NTD];
  const int tid = threadIdx.x;
  const int b  = blockIdx.x / (Nn / NTD);
  const int n0 = (blockIdx.x % (Nn / NTD)) * NTD;

  if (tid < NTD) scodes[tid] = codes[(size_t)b * Nn + n0 + tid];
  __syncthreads();

  for (int r = 0; r < NTD; ++r)           // coalesced gather of code rows
    rows[r][tid] = e[(size_t)scodes[r] * Dd + tid];
  __syncthreads();

  #pragma unroll 4
  for (int it = 0; it < NTD; ++it) {      // coalesced transposed write
    const int i = tid & 31;
    const int d = it * 8 + (tid >> 5);
    out[((size_t)b * Dd + d) * Nn + n0 + i] = rows[i][d];
  }
}

// ---------------------------------------------------------------------------
extern "C" void kernel_launch(void* const* d_in, const int* in_sizes, int n_in,
                              void* d_out, int out_size, void* d_ws, size_t ws_size,
                              hipStream_t stream) {
  const float* x = (const float*)d_in[0];   // [B, D, N]
  const float* e = (const float*)d_in[1];   // [K, D]
  float* out = (float*)d_out;               // [B, D, N]

  char* ws = (char*)d_ws;
  __bf16* ebf   = (__bf16*)ws;                                  // 4 MB
  float*  e2w   = (float*)(ws + (size_t)Kk * Dd * sizeof(__bf16));
  int*    codes = (int*)(ws + (size_t)Kk * Dd * sizeof(__bf16)
                            + (size_t)Kk * sizeof(float));

  vq_prep_kernel<<<Kk, 256, 0, stream>>>(e, ebf, e2w);
  vq_argmin_kernel<<<Bb * (Nn / NT), 256, 0, stream>>>(x, ebf, e2w, codes);
  vq_decode_kernel<<<Bb * (Nn / 32), 256, 0, stream>>>(e, codes, out);
}
